// SEAN_Block_16157666967824
// MI455X (gfx1250) — matrix operand order repaired and checked by reading, not compiled, not run on hardware
//
#include <hip/hip_runtime.h>
#include <hip/hip_bf16.h>

// ---------------------------------------------------------------------------
// SEAN block for MI455X (gfx1250, wave32, WMMA bf16 16x16x32)
//
// Algebra: style convs folded into per-batch effective 19-channel conv
// weights; sigmoid mixing weights baked into conv weights so gamma-path +
// style-path accumulate into one WMMA accumulator; final kernel applies
// instance norm and writes the output directly (no gamma/beta intermediates).
//
// Data movement: im2col tiles are materialized in LDS in the WMMA B-fragment
// native layout (pixel-major, K contiguous) so B fragments are 2x ds_load_b128
// per lane; A fragments are 2x global_load_b128 per lane from packed bf16
// weights (L2-resident).
// ---------------------------------------------------------------------------

typedef __bf16 bf16_t;
typedef __attribute__((ext_vector_type(16))) __bf16 v16bf;
typedef __attribute__((ext_vector_type(8)))  __bf16 v8bf;
typedef __attribute__((ext_vector_type(8)))  float  v8f;

#define BATCH      8
#define NORM_NC    256
#define LABEL_NC   19
#define STYLE_N    512
#define NHIDDEN    128
#define IMG_H      128
#define IMG_W      128
#define KSEG       171      // 19*9
#define KSEG_PAD   192      // pad to multiple of 32
#define KACT       1152     // 128*9
#define KCH        288      // actv K-chunk: 32 channels * 9
#define PITCH_ACT  296      // KCH + 8 halves (bank spread, 16B aligned rows)
#define PITCH_SEG  200      // KSEG_PAD + 8 halves

// ---- workspace layout (bytes) ----
static constexpr size_t WS_STATS  = 0;                                        // 2048*2 f32
static constexpr size_t WS_BIAS_S = WS_STATS + (size_t)2048*2*4;              // 256 f32
static constexpr size_t WS_BIAS_O = WS_BIAS_S + 256*4;                        // 256 f32
static constexpr size_t WS_SEG    = 32768;                                    // bf16 8*19*128*128
static constexpr size_t WS_WSH    = WS_SEG + (size_t)BATCH*LABEL_NC*IMG_H*IMG_W*2;
static constexpr size_t WS_WG     = WS_WSH + (size_t)NHIDDEN*KSEG_PAD*2;
static constexpr size_t WS_WB     = WS_WG  + (size_t)NORM_NC*KACT*2;
static constexpr size_t WS_SG     = WS_WB  + (size_t)NORM_NC*KACT*2;
static constexpr size_t WS_SB     = WS_SG  + (size_t)BATCH*NORM_NC*KSEG_PAD*2;
static constexpr size_t WS_ACT    = WS_SB  + (size_t)BATCH*NORM_NC*KSEG_PAD*2;

__device__ __forceinline__ float sigmoidf_(float a) { return 1.0f / (1.0f + __expf(-a)); }

// A-matrix fragment (16x32 bf16), ISA layout:
//   lane L: m = L%16; halves 0..7 = K[(L/16)*8 .. +7]; halves 8..15 = K[16+(L/16)*8 .. +7]
__device__ __forceinline__ v16bf load_a_frag(const bf16_t* __restrict__ W, int stride,
                                             int o0, int kbase, int lane) {
    int m  = lane & 15;
    int kb = (lane >> 4) << 3;
    const bf16_t* p = W + (size_t)(o0 + m) * stride + kbase + kb;
    v8bf lo = *reinterpret_cast<const v8bf*>(p);
    v8bf hi = *reinterpret_cast<const v8bf*>(p + 16);
    v16bf a;
#pragma unroll
    for (int j = 0; j < 8; ++j) { a[j] = lo[j]; a[8 + j] = hi[j]; }
    return a;
}

// B-matrix fragment (32x16 bf16) from an LDS im2col tile stored B[n][k]:
//   lane L: n = L%16 (+tile base); halves 0..7 = K[kbase + (L/16)*8 .. +7];
//   halves 8..15 = K[kbase + 16 + (L/16)*8 .. +7] = two ds_load_b128.
template<int PITCH>
__device__ __forceinline__ v16bf load_b_im2col(const bf16_t* __restrict__ lds,
                                               int n0, int kbase, int lane) {
    const bf16_t* p = lds + (n0 + (lane & 15)) * PITCH + kbase + ((lane >> 4) << 3);
    v8bf lo = *reinterpret_cast<const v8bf*>(p);
    v8bf hi = *reinterpret_cast<const v8bf*>(p + 16);
    v16bf b;
#pragma unroll
    for (int j = 0; j < 8; ++j) { b[j] = lo[j]; b[8 + j] = hi[j]; }
    return b;
}

// Build im2col tile B[n][k] (n in [0,32), k in [0,KCHUNK)) from planar bf16
// src[b][c][H][W]. k decodes as (c0 + k/9, ky, kx). Boundary -> 0.
// Consecutive threads take consecutive n -> coalesced global reads, bank-
// spread LDS writes.
template<int KCHUNK, int PITCH>
__device__ __forceinline__ void build_im2col(const bf16_t* __restrict__ src, int nCh,
                                             int c0, int b, int y, int x0,
                                             bf16_t* __restrict__ dst, int tid) {
    for (int idx = tid; idx < 32 * KCHUNK; idx += 256) {
        int k  = idx >> 5;
        int n  = idx & 31;
        int cq = k / 9;
        int r  = k - cq * 9;
        int ky = r / 3, kx = r - ky * 3;
        int c  = c0 + cq;
        int gy = y - 1 + ky, gx = x0 - 1 + n + kx;
        bf16_t v = (bf16_t)0.0f;
        if (c < nCh && (unsigned)gy < 128u && (unsigned)gx < 128u)
            v = src[(((size_t)b * nCh + c) * IMG_H + gy) * IMG_W + gx];
        dst[n * PITCH + k] = v;
    }
}

// ---------------------------------------------------------------------------
// 1) Instance-norm statistics: one block per (b, c) row of 16384 pixels.
__global__ void __launch_bounds__(256) stats_kernel(const float* __restrict__ x,
                                                    float* __restrict__ stats) {
    __shared__ float s1[256], s2v[256];
    int row = blockIdx.x;   // b*256 + c
    const float* p = x + (size_t)row * (IMG_H * IMG_W);
    float s = 0.f, q = 0.f;
    for (int i = threadIdx.x; i < IMG_H * IMG_W; i += 256) {
        float v = p[i]; s += v; q += v * v;
    }
    s1[threadIdx.x] = s; s2v[threadIdx.x] = q;
    __syncthreads();
    for (int off = 128; off > 0; off >>= 1) {
        if ((int)threadIdx.x < off) {
            s1[threadIdx.x]  += s1[threadIdx.x + off];
            s2v[threadIdx.x] += s2v[threadIdx.x + off];
        }
        __syncthreads();
    }
    if (threadIdx.x == 0) {
        float mean = s1[0] * (1.0f / 16384.0f);
        float var  = s2v[0] * (1.0f / 16384.0f) - mean * mean;
        stats[row * 2]     = mean;
        stats[row * 2 + 1] = rsqrtf(var + 1e-5f);
    }
}

// 2) segmap f32 -> bf16
__global__ void __launch_bounds__(256) seg_bf16_kernel(const float* __restrict__ seg,
                                                       bf16_t* __restrict__ out, int n) {
    int i = blockIdx.x * 256 + threadIdx.x;
    if (i < n) out[i] = (bf16_t)seg[i];
}

// 3) pack w_shared [128][171] -> bf16 [128][192] (zero padded K)
__global__ void __launch_bounds__(256) pack_shared_kernel(const float* __restrict__ w,
                                                          bf16_t* __restrict__ out) {
    int i = blockIdx.x * 256 + threadIdx.x;
    if (i >= NHIDDEN * KSEG_PAD) return;
    int o = i / KSEG_PAD, k = i - o * KSEG_PAD;
    out[i] = (k < KSEG) ? (bf16_t)w[o * KSEG + k] : (bf16_t)0.0f;
}

// 4) pack w_gamma/w_beta scaled by (1-sigmoid(alpha)) -> bf16 [256][1152]
__global__ void __launch_bounds__(256) pack_big_kernel(const float* __restrict__ wg,
                                                       const float* __restrict__ wb,
                                                       const float* __restrict__ ag,
                                                       const float* __restrict__ ab,
                                                       bf16_t* __restrict__ outg,
                                                       bf16_t* __restrict__ outb) {
    int i = blockIdx.x * 256 + threadIdx.x;
    if (i >= NORM_NC * KACT) return;
    float sg = 1.0f - sigmoidf_(ag[0]);
    float sb = 1.0f - sigmoidf_(ab[0]);
    outg[i] = (bf16_t)(sg * wg[i]);
    outb[i] = (bf16_t)(sb * wb[i]);
}

// 5) per-batch effective style weights, pre-scaled by sigmoid(alpha):
//    out[b][o][l*9+kk] = sig(a) * sum_s w_s[o,s,kk] * style[b,l,s]
__global__ void __launch_bounds__(256) build_style_kernel(const float* __restrict__ style,
                                                          const float* __restrict__ wsg,
                                                          const float* __restrict__ wsb,
                                                          const float* __restrict__ ag,
                                                          const float* __restrict__ ab,
                                                          bf16_t* __restrict__ outg,
                                                          bf16_t* __restrict__ outb) {
    int i = blockIdx.x * 256 + threadIdx.x;   // over BATCH*NORM_NC*KSEG_PAD
    if (i >= BATCH * NORM_NC * KSEG_PAD) return;
    int k  = i % KSEG_PAD;
    int bo = i / KSEG_PAD;
    int o  = bo % NORM_NC;
    int b  = bo / NORM_NC;
    if (k >= KSEG) { outg[i] = (bf16_t)0.0f; outb[i] = (bf16_t)0.0f; return; }
    int l = k / 9, kk = k - l * 9;
    const float* st = style + ((size_t)b * LABEL_NC + l) * STYLE_N;
    const float* pg = wsg + (size_t)o * STYLE_N * 9 + kk;
    const float* pb = wsb + (size_t)o * STYLE_N * 9 + kk;
    float dg = 0.f, db = 0.f;
    for (int s = 0; s < STYLE_N; ++s) {
        float sv = st[s];
        dg += sv * pg[s * 9];
        db += sv * pb[s * 9];
    }
    outg[i] = (bf16_t)(sigmoidf_(ag[0]) * dg);
    outb[i] = (bf16_t)(sigmoidf_(ab[0]) * db);
}

// 6) fused bias vectors (scale bias includes the trailing +1.0)
__global__ void __launch_bounds__(256) bias_kernel(const float* __restrict__ bg,
                                                   const float* __restrict__ bb,
                                                   const float* __restrict__ bsg,
                                                   const float* __restrict__ bsb,
                                                   const float* __restrict__ ag,
                                                   const float* __restrict__ ab,
                                                   float* __restrict__ bias_s,
                                                   float* __restrict__ bias_o) {
    int o = threadIdx.x;
    float wgv = sigmoidf_(ag[0]), wbv = sigmoidf_(ab[0]);
    bias_s[o] = wgv * bsg[o] + (1.0f - wgv) * bg[o] + 1.0f;
    bias_o[o] = wbv * bsb[o] + (1.0f - wbv) * bb[o];
}

// ---------------------------------------------------------------------------
// 7) actv = relu(conv(segmap, w_shared) + b_shared), bf16 planar [b][c][h][w]
//    One block per (b, row, 32-px strip). im2col in LDS, K = 192.
__global__ void __launch_bounds__(256) actv_kernel(const bf16_t* __restrict__ segb,
                                                   const bf16_t* __restrict__ wsh,
                                                   const float* __restrict__ bsh,
                                                   bf16_t* __restrict__ actv) {
    __shared__ bf16_t Bseg[32 * PITCH_SEG];
    int bid = blockIdx.x;
    int strip = bid & 3, y = (bid >> 2) & 127, b = bid >> 9;
    int x0 = strip * 32;
    int tid = threadIdx.x, lane = tid & 31, wid = tid >> 5;

    build_im2col<KSEG_PAD, PITCH_SEG>(segb, LABEL_NC, 0, b, y, x0, Bseg, tid);
    __syncthreads();

    v8f acc[2];
#pragma unroll
    for (int ti = 0; ti < 2; ++ti)
#pragma unroll
        for (int r = 0; r < 8; ++r) acc[ti][r] = 0.f;

#pragma unroll
    for (int ti = 0; ti < 2; ++ti) {
        int t = wid + 8 * ti;            // 2 n-tiles x 8 m-tiles
        int nt = t & 1, mt = t >> 1;
#pragma unroll
        for (int ks = 0; ks < 6; ++ks) {
            v16bf bfr = load_b_im2col<PITCH_SEG>(Bseg, nt * 16, ks * 32, lane);
            v16bf a   = load_a_frag(wsh, KSEG_PAD, mt * 16, ks * 32, lane);
            acc[ti] = __builtin_amdgcn_wmma_f32_16x16x32_bf16(false, a, false, bfr,
                                                              (short)0, acc[ti], false, false);
        }
    }

    int n = lane & 15, mhalf = (lane >> 4) * 8;
#pragma unroll
    for (int ti = 0; ti < 2; ++ti) {
        int t = wid + 8 * ti;
        int nt = t & 1, mt = t >> 1;
        int x = x0 + nt * 16 + n;
#pragma unroll
        for (int r = 0; r < 8; ++r) {
            int o = mt * 16 + r + mhalf;
            float v = acc[ti][r] + bsh[o];
            v = v > 0.f ? v : 0.f;
            actv[(((size_t)b * NHIDDEN + o) * IMG_H + y) * IMG_W + x] = (bf16_t)v;
        }
    }
}

// ---------------------------------------------------------------------------
// 8) Mega kernel: scale = conv(actv,(1-wg)w_gamma)+conv(segmap,wg*Weff_g)+bias_s
//                 offs  = conv(actv,(1-wb)w_beta )+conv(segmap,wb*Weff_b)+bias_o
//                 out   = (x - mean)*invstd*scale + offs
//    K over actv processed in 4 chunks of 288 (32ch x 9) staged as LDS im2col;
//    each B fragment feeds both the scale and the offset accumulators.
__global__ void __launch_bounds__(256) main_kernel(const float* __restrict__ x,
                                                   const bf16_t* __restrict__ segb,
                                                   const bf16_t* __restrict__ actv,
                                                   const bf16_t* __restrict__ wgp,
                                                   const bf16_t* __restrict__ wbp,
                                                   const bf16_t* __restrict__ wsg,
                                                   const bf16_t* __restrict__ wsb,
                                                   const float* __restrict__ stats,
                                                   const float* __restrict__ bias_s,
                                                   const float* __restrict__ bias_o,
                                                   float* __restrict__ out) {
    __shared__ bf16_t Bact[32 * PITCH_ACT];
    __shared__ bf16_t Bseg[32 * PITCH_SEG];
    int bid = blockIdx.x;
    int strip = bid & 3, y = (bid >> 2) & 127, b = bid >> 9;
    int x0 = strip * 32;
    int tid = threadIdx.x, lane = tid & 31, wid = tid >> 5;

    build_im2col<KSEG_PAD, PITCH_SEG>(segb, LABEL_NC, 0, b, y, x0, Bseg, tid);
    __syncthreads();

    const bf16_t* wsg_b = wsg + (size_t)b * NORM_NC * KSEG_PAD;
    const bf16_t* wsb_b = wsb + (size_t)b * NORM_NC * KSEG_PAD;

    // per-wave accumulators: 4 tiles x (scale, offset)
    v8f accS[4], accO[4];
#pragma unroll
    for (int ti = 0; ti < 4; ++ti)
#pragma unroll
        for (int r = 0; r < 8; ++r) { accS[ti][r] = 0.f; accO[ti][r] = 0.f; }

    // ---- style path (K = 192 over segmap) ----
#pragma unroll
    for (int ti = 0; ti < 4; ++ti) {
        int t = wid + 8 * ti;            // 2 n-tiles x 16 m-tiles
        int nt = t & 1, mt = t >> 1, o0 = mt * 16;
#pragma unroll
        for (int ks = 0; ks < 6; ++ks) {
            v16bf bfr = load_b_im2col<PITCH_SEG>(Bseg, nt * 16, ks * 32, lane);
            v16bf ag  = load_a_frag(wsg_b, KSEG_PAD, o0, ks * 32, lane);
            accS[ti] = __builtin_amdgcn_wmma_f32_16x16x32_bf16(false, ag, false, bfr,
                                                               (short)0, accS[ti], false, false);
            v16bf ab  = load_a_frag(wsb_b, KSEG_PAD, o0, ks * 32, lane);
            accO[ti] = __builtin_amdgcn_wmma_f32_16x16x32_bf16(false, ab, false, bfr,
                                                               (short)0, accO[ti], false, false);
        }
    }

    // ---- actv path (K = 1152 in 4 chunks of 288) ----
    for (int kc = 0; kc < 4; ++kc) {
        __syncthreads();   // protect previous chunk's readers
        build_im2col<KCH, PITCH_ACT>(actv, NHIDDEN, kc * 32, b, y, x0, Bact, tid);
        __syncthreads();
#pragma unroll
        for (int ti = 0; ti < 4; ++ti) {
            int t = wid + 8 * ti;
            int nt = t & 1, mt = t >> 1, o0 = mt * 16;
#pragma unroll
            for (int ks = 0; ks < 9; ++ks) {
                v16bf bfr = load_b_im2col<PITCH_ACT>(Bact, nt * 16, ks * 32, lane);
                v16bf ag  = load_a_frag(wgp, KACT, o0, kc * KCH + ks * 32, lane);
                accS[ti] = __builtin_amdgcn_wmma_f32_16x16x32_bf16(false, ag, false, bfr,
                                                                   (short)0, accS[ti], false, false);
                v16bf ab  = load_a_frag(wbp, KACT, o0, kc * KCH + ks * 32, lane);
                accO[ti] = __builtin_amdgcn_wmma_f32_16x16x32_bf16(false, ab, false, bfr,
                                                                   (short)0, accO[ti], false, false);
            }
        }
    }

    // ---- finalize: instance norm + fused scale/offset ----
    int n = lane & 15, mhalf = (lane >> 4) * 8;
#pragma unroll
    for (int ti = 0; ti < 4; ++ti) {
        int t = wid + 8 * ti;
        int nt = t & 1, mt = t >> 1, o0 = mt * 16;
        int xg = x0 + nt * 16 + n;
#pragma unroll
        for (int r = 0; r < 8; ++r) {
            int o = o0 + r + mhalf;
            size_t pix = (((size_t)b * NORM_NC + o) * IMG_H + y) * IMG_W + xg;
            float mean = stats[(b * NORM_NC + o) * 2];
            float inv  = stats[(b * NORM_NC + o) * 2 + 1];
            float scale = accS[ti][r] + bias_s[o];
            float offv  = accO[ti][r] + bias_o[o];
            out[pix] = (x[pix] - mean) * inv * scale + offv;
        }
    }
}

// ---------------------------------------------------------------------------
extern "C" void kernel_launch(void* const* d_in, const int* in_sizes, int n_in,
                              void* d_out, int out_size, void* d_ws, size_t ws_size,
                              hipStream_t stream) {
    const float* x         = (const float*)d_in[0];
    const float* segmap    = (const float*)d_in[1];
    const float* style     = (const float*)d_in[2];
    const float* w_shared  = (const float*)d_in[3];
    const float* b_shared  = (const float*)d_in[4];
    const float* w_gamma   = (const float*)d_in[5];
    const float* b_gamma   = (const float*)d_in[6];
    const float* w_beta    = (const float*)d_in[7];
    const float* b_beta    = (const float*)d_in[8];
    const float* w_sgamma  = (const float*)d_in[9];
    const float* b_sgamma  = (const float*)d_in[10];
    const float* w_sbeta   = (const float*)d_in[11];
    const float* b_sbeta   = (const float*)d_in[12];
    const float* alpha_b   = (const float*)d_in[13];
    const float* alpha_g   = (const float*)d_in[14];

    char* ws = (char*)d_ws;
    float*  stats  = (float*) (ws + WS_STATS);
    float*  bias_s = (float*) (ws + WS_BIAS_S);
    float*  bias_o = (float*) (ws + WS_BIAS_O);
    bf16_t* segb   = (bf16_t*)(ws + WS_SEG);
    bf16_t* wshp   = (bf16_t*)(ws + WS_WSH);
    bf16_t* wgp    = (bf16_t*)(ws + WS_WG);
    bf16_t* wbp    = (bf16_t*)(ws + WS_WB);
    bf16_t* wsgp   = (bf16_t*)(ws + WS_SG);
    bf16_t* wsbp   = (bf16_t*)(ws + WS_SB);
    bf16_t* actvp  = (bf16_t*)(ws + WS_ACT);

    const int SEGN = BATCH * LABEL_NC * IMG_H * IMG_W;

    stats_kernel<<<BATCH * NORM_NC, 256, 0, stream>>>(x, stats);
    seg_bf16_kernel<<<(SEGN + 255) / 256, 256, 0, stream>>>(segmap, segb, SEGN);
    pack_shared_kernel<<<(NHIDDEN * KSEG_PAD + 255) / 256, 256, 0, stream>>>(w_shared, wshp);
    pack_big_kernel<<<(NORM_NC * KACT + 255) / 256, 256, 0, stream>>>(
        w_gamma, w_beta, alpha_g, alpha_b, wgp, wbp);
    build_style_kernel<<<(BATCH * NORM_NC * KSEG_PAD + 255) / 256, 256, 0, stream>>>(
        style, w_sgamma, w_sbeta, alpha_g, alpha_b, wsgp, wsbp);
    bias_kernel<<<1, 256, 0, stream>>>(b_gamma, b_beta, b_sgamma, b_sbeta,
                                       alpha_g, alpha_b, bias_s, bias_o);
    actv_kernel<<<BATCH * IMG_H * 4, 256, 0, stream>>>(segb, wshp, b_shared, actvp);
    main_kernel<<<BATCH * IMG_H * 4, 256, 0, stream>>>(
        x, segb, actvp, wgp, wbp, wsgp, wsbp, stats, bias_s, bias_o, (float*)d_out);
}
